// WorldModel_19138374271050
// MI455X (gfx1250) — compile-verified, matrix operand order checked by reading, not run on hardware
//
#include <hip/hip_runtime.h>
#include <hip/hip_bf16.h>
#include <stdint.h>

// ---------------- problem constants ----------------
#define STOCH   32
#define ACTD    8
#define OBSD    64
#define HIDD    512
#define DETERD  512
#define BSZ     1024
#define TLEN    64
#define OUTD    672          // q_mean(32) q_std(32) stoch(32) p_mean(32) p_std(32) deter(512)
#define MIN_STD 0.1f
#define UPD_BIAS (-1.0f)
#define LN_EPS  1e-3f

// ---------------- tiling ----------------
#define BM      16           // batch rows per block
#define NTH     256          // 8 wave32
#define NWAVES  8
#define GS      1536         // gbuf row stride (f32)
#define AS      1024         // abuf row stride (bf16/ushort)

typedef __attribute__((ext_vector_type(16))) __bf16   v16bf;
typedef __attribute__((ext_vector_type(8)))  float    v8f;
typedef __attribute__((ext_vector_type(4)))  float    f32x4;
typedef __attribute__((ext_vector_type(2)))  float    f32x2;
typedef __attribute__((ext_vector_type(4)))  unsigned u32x4;

union FragBF { u32x4 q[2]; v16bf v; };

__device__ __forceinline__ unsigned short f2bf(float f) {
  unsigned u = __float_as_uint(f);
  u += 0x7fffu + ((u >> 16) & 1u);           // round-to-nearest-even
  return (unsigned short)(u >> 16);
}
__device__ __forceinline__ unsigned pack2bf(float a, float b) {
  return (unsigned)f2bf(a) | ((unsigned)f2bf(b) << 16);
}
__device__ __forceinline__ float sigm(float x)  { return 1.f / (1.f + __expf(-x)); }
__device__ __forceinline__ float siluf(float x) { return x * sigm(x); }
__device__ __forceinline__ float softplusf(float x) {
  return fmaxf(x, 0.f) + log1pf(__expf(-fabsf(x)));
}

// Warm the head of the next GEMM's weight stream from L2 while VALU phases run.
__device__ __forceinline__ void wprefetch(const unsigned short* Wt, int bytes, int tid) {
  int off = tid * 128;                       // one cacheline per thread
  if (off < bytes) __builtin_prefetch((const char*)Wt + off, 0, 3);
}

// ---------------------------------------------------------------------------
// WMMA GEMM: A (16 x KP bf16, LDS, row stride AS) x Wt (N-major bf16 weights,
// contiguous K runs) -> gbuf (f32 LDS, stride GS).  Two N-tiles per wave per
// pass share one A fragment (halves LDS A traffic, back-to-back WMMA issue).
// `wave` is SGPR-uniform (readfirstlane) so tile loops are scalar branches and
// EXEC stays all-ones around every WMMA, per ISA 7.12 restrictions.
// Fragment layouts per CDNA5 ISA 7.12.2 (16-bit A 16x32; 16x16 f32 C).
// ---------------------------------------------------------------------------
template <int KP>
__device__ __forceinline__ void wmma_gemm(const unsigned short* __restrict__ Wt,
                                          int Ntiles, const unsigned short* abuf,
                                          float* gbuf, int lane, int wave) {
  const int nl  = lane & 15;      // A row (M) / B column within tile
  const int hlf = lane >> 4;      // lane half selects K sub-block
  const unsigned short* arow = abuf + nl * AS;
  for (int nt0 = wave; nt0 < Ntiles; nt0 += 2 * NWAVES) {
    const int nt1 = nt0 + NWAVES;
    const unsigned short* w0 = Wt + (size_t)(nt0 * 16 + nl) * KP;
    v8f acc0 = {0.f, 0.f, 0.f, 0.f, 0.f, 0.f, 0.f, 0.f};
    if (nt1 < Ntiles) {
      const unsigned short* w1 = Wt + (size_t)(nt1 * 16 + nl) * KP;
      v8f acc1 = {0.f, 0.f, 0.f, 0.f, 0.f, 0.f, 0.f, 0.f};
#pragma unroll 4
      for (int kb = 0; kb < KP; kb += 32) {
        FragBF fa, fb0, fb1;
        fa.q[0]  = *(const u32x4*)(arow + kb + 8 * hlf);
        fa.q[1]  = *(const u32x4*)(arow + kb + 16 + 8 * hlf);
        fb0.q[0] = *(const u32x4*)(w0 + kb + 8 * hlf);
        fb0.q[1] = *(const u32x4*)(w0 + kb + 16 + 8 * hlf);
        fb1.q[0] = *(const u32x4*)(w1 + kb + 8 * hlf);
        fb1.q[1] = *(const u32x4*)(w1 + kb + 16 + 8 * hlf);
        acc0 = __builtin_amdgcn_wmma_f32_16x16x32_bf16(false, fa.v, false, fb0.v,
                                                       (short)0, acc0, false, false);
        acc1 = __builtin_amdgcn_wmma_f32_16x16x32_bf16(false, fa.v, false, fb1.v,
                                                       (short)0, acc1, false, false);
      }
#pragma unroll
      for (int i = 0; i < 8; i++) {
        gbuf[(size_t)(i + 8 * hlf) * GS + nt0 * 16 + nl] = acc0[i];
        gbuf[(size_t)(i + 8 * hlf) * GS + nt1 * 16 + nl] = acc1[i];
      }
    } else {
#pragma unroll 4
      for (int kb = 0; kb < KP; kb += 32) {
        FragBF fa, fb0;
        fa.q[0]  = *(const u32x4*)(arow + kb + 8 * hlf);
        fa.q[1]  = *(const u32x4*)(arow + kb + 16 + 8 * hlf);
        fb0.q[0] = *(const u32x4*)(w0 + kb + 8 * hlf);
        fb0.q[1] = *(const u32x4*)(w0 + kb + 16 + 8 * hlf);
        acc0 = __builtin_amdgcn_wmma_f32_16x16x32_bf16(false, fa.v, false, fb0.v,
                                                       (short)0, acc0, false, false);
      }
#pragma unroll
      for (int i = 0; i < 8; i++)
        gbuf[(size_t)(i + 8 * hlf) * GS + nt0 * 16 + nl] = acc0[i];
    }
  }
}

// Row mean / rstd over gbuf[16][N] -> stats[16][2].  256 threads: 16 per row,
// float4 LDS reads.
__device__ __forceinline__ void row_stats(const float* gbuf, int N,
                                          float* red, float* stats, int tid) {
  __syncthreads();                       // gbuf produced by WMMA stores
  const int row = tid >> 4, sub = tid & 15;
  float s = 0.f, ss = 0.f;
  for (int j = sub * 4; j < N; j += 64) {
    f32x4 x = *(const f32x4*)(gbuf + row * GS + j);
    s  += x.x + x.y + x.z + x.w;
    ss += x.x * x.x + x.y * x.y + x.z * x.z + x.w * x.w;
  }
  red[(row * 16 + sub) * 2 + 0] = s;
  red[(row * 16 + sub) * 2 + 1] = ss;
  __syncthreads();
  if (sub == 0) {
    float S = 0.f, SS = 0.f;
#pragma unroll
    for (int i = 0; i < 16; i++) {
      S  += red[(row * 16 + i) * 2];
      SS += red[(row * 16 + i) * 2 + 1];
    }
    float mean = S / (float)N;
    float var  = SS / (float)N - mean * mean;
    stats[row * 2 + 0] = mean;
    stats[row * 2 + 1] = rsqrtf(var + LN_EPS);
  }
  __syncthreads();
}

// ---------------------------------------------------------------------------
// Prep: W (Kreal x N, f32, row-major) -> Wt (N x Kp, bf16, zero K-padding)
// ---------------------------------------------------------------------------
__global__ void k_wprep(const float* __restrict__ W, unsigned short* __restrict__ Wt,
                        int Kreal, int Kp, int N) {
  int total = N * Kp;
  for (int idx = blockIdx.x * blockDim.x + threadIdx.x; idx < total;
       idx += gridDim.x * blockDim.x) {
    int n = idx / Kp, k = idx - n * Kp;
    float v = (k < Kreal) ? W[(size_t)k * N + n] : 0.f;
    Wt[idx] = f2bf(v);
  }
}

// ---------------------------------------------------------------------------
// Prep: initial state.  d_init = tanh(learned);  s_init = prior mean at d_init.
// ---------------------------------------------------------------------------
__global__ void k_init_state(const float* __restrict__ init_deter,
                             const float* __restrict__ W_po, const float* __restrict__ g_po,
                             const float* __restrict__ b_po,
                             const float* __restrict__ W_ps, const float* __restrict__ b_ps,
                             float* __restrict__ d_init, float* __restrict__ s_init) {
  __shared__ float dsh[DETERD];
  __shared__ float hsh[HIDD];
  __shared__ float mr[2];
  int tid = threadIdx.x;
  float d = tanhf(init_deter[tid]);
  dsh[tid] = d;
  d_init[tid] = d;
  __syncthreads();
  float acc = 0.f;
  for (int k = 0; k < DETERD; k++) acc += dsh[k] * W_po[(size_t)k * HIDD + tid];
  hsh[tid] = acc;
  __syncthreads();
  if (tid == 0) {
    float S = 0.f, SS = 0.f;
    for (int i = 0; i < HIDD; i++) { S += hsh[i]; SS += hsh[i] * hsh[i]; }
    float mean = S / (float)HIDD;
    mr[0] = mean;
    mr[1] = rsqrtf(SS / (float)HIDD - mean * mean + LN_EPS);
  }
  __syncthreads();
  float y = (acc - mr[0]) * mr[1] * g_po[tid] + b_po[tid];
  hsh[tid] = siluf(y);
  __syncthreads();
  if (tid < STOCH) {
    float s = b_ps[tid];
    for (int j = 0; j < HIDD; j++) s += hsh[j] * W_ps[(size_t)j * (2 * STOCH) + tid];
    s_init[tid] = s;
  }
}

// ---------------------------------------------------------------------------
// Main RSSM scan: each block owns BM batch rows for all T steps (no grid sync
// needed: the recurrence is independent per batch row).
// ---------------------------------------------------------------------------
__global__ void __launch_bounds__(NTH) k_rssm(
    const float* __restrict__ obs, const float* __restrict__ action,
    const int* __restrict__ is_first,
    const float* __restrict__ g_pi, const float* __restrict__ b_pi,
    const float* __restrict__ g_gru, const float* __restrict__ b_gru,
    const float* __restrict__ g_po, const float* __restrict__ b_po,
    const float* __restrict__ b_ps,
    const float* __restrict__ g_pn, const float* __restrict__ b_pn,
    const float* __restrict__ b_pt,
    const unsigned short* __restrict__ Wt_pi, const unsigned short* __restrict__ Wt_gru,
    const unsigned short* __restrict__ Wt_po, const unsigned short* __restrict__ Wt_ps,
    const unsigned short* __restrict__ Wt_pn, const unsigned short* __restrict__ Wt_pt,
    const float* __restrict__ d_init, const float* __restrict__ s_init,
    float* __restrict__ out) {

  extern __shared__ char smem[];
  float* deter = (float*)smem;                       // 16*512
  float* stoch = deter + BM * DETERD;                // 16*32
  float* gbuf  = stoch + BM * STOCH;                 // 16*1536
  float* red   = gbuf + BM * GS;                     // 512
  float* stats = red + 512;                          // 32
  unsigned short* abuf = (unsigned short*)(stats + 32);  // 16*1024 bf16 (16B aligned)

  const int tid  = threadIdx.x;
  const int lane = tid & 31;
  const int wave = __builtin_amdgcn_readfirstlane(tid >> 5);  // SGPR-uniform wave id
  const int r0   = blockIdx.x * BM;

  // carry init: (d_init, s_init) for every row
  for (int m = 0; m < BM; m++) {
    for (int j = tid; j < DETERD; j += NTH) deter[m * DETERD + j] = d_init[j];
    for (int j = tid; j < STOCH;  j += NTH) stoch[m * STOCH + j]  = s_init[j];
  }
  __syncthreads();

  for (int t = 0; t < TLEN; t++) {
    // ---- (1) is_first reset + build A0 = [stoch | act | pad] (16 x 64 bf16) ----
    wprefetch(Wt_pi, 64 * 512 * 2, tid);
    for (int m = 0; m < BM; m++) {
      const int f = is_first[(size_t)(r0 + m) * TLEN + t];
      for (int j = tid; j < DETERD; j += NTH)
        if (f) deter[m * DETERD + j] = d_init[j];
      for (int j = tid; j < STOCH; j += NTH) {
        float s = f ? s_init[j] : stoch[m * STOCH + j];
        stoch[m * STOCH + j] = s;
        abuf[m * AS + j] = f2bf(s);
      }
      for (int j = tid; j < ACTD; j += NTH) {
        float a = f ? 0.f
                    : __builtin_nontemporal_load(
                          action + ((size_t)(r0 + m) * TLEN + t) * ACTD + j);
        abuf[m * AS + STOCH + j] = f2bf(a);
      }
      for (int j = tid; j < 24; j += NTH) abuf[m * AS + 40 + j] = 0;  // pad 40->64
    }
    __syncthreads();

    // ---- (2) prior_in: x = silu(LN(A0 @ W_pi)) ----
    wmma_gemm<64>(Wt_pi, HIDD / 16, abuf, gbuf, lane, wave);
    wprefetch(Wt_gru, 1024 * 1536 * 2, tid);
    row_stats(gbuf, HIDD, red, stats, tid);
    for (int m = 0; m < BM; m++) {
      const float mean = stats[2 * m], rstd = stats[2 * m + 1];
      for (int jj = tid; jj < HIDD / 2; jj += NTH) {
        const int j = jj * 2;
        f32x2 x = *(const f32x2*)(gbuf + m * GS + j);
        float y0 = siluf((x.x - mean) * rstd * g_pi[j] + b_pi[j]);
        float y1 = siluf((x.y - mean) * rstd * g_pi[j + 1] + b_pi[j + 1]);
        *(unsigned*)(abuf + m * AS + j) = pack2bf(y0, y1);        // x -> A[:,0:512]
      }
      for (int jj = tid; jj < DETERD / 2; jj += NTH) {
        const int j = jj * 2;
        f32x2 d = *(const f32x2*)(deter + m * DETERD + j);
        *(unsigned*)(abuf + m * AS + HIDD + j) = pack2bf(d.x, d.y);  // deter -> A[:,512:]
      }
    }
    __syncthreads();

    // ---- (3) GRU: gates = LN([x|deter] @ W_gru); blend deter ----
    wmma_gemm<HIDD + DETERD>(Wt_gru, (3 * DETERD) / 16, abuf, gbuf, lane, wave);
    wprefetch(Wt_po, 512 * 512 * 2, tid);
    row_stats(gbuf, 3 * DETERD, red, stats, tid);
    for (int m = 0; m < BM; m++) {
      const float mean = stats[2 * m], rstd = stats[2 * m + 1];
      for (int jj = tid; jj < (3 * DETERD) / 2; jj += NTH) {
        const int j = jj * 2;
        f32x2 x = *(const f32x2*)(gbuf + m * GS + j);
        x.x = (x.x - mean) * rstd * g_gru[j] + b_gru[j];
        x.y = (x.y - mean) * rstd * g_gru[j + 1] + b_gru[j + 1];
        *(f32x2*)(gbuf + m * GS + j) = x;
      }
    }
    __syncthreads();
    for (int m = 0; m < BM; m++) {
      const size_t ob = ((size_t)(r0 + m) * TLEN + t) * OUTD;
      for (int jj = tid; jj < DETERD / 2; jj += NTH) {
        const int j = jj * 2;
        f32x2 rg = *(const f32x2*)(gbuf + m * GS + j);
        f32x2 cg = *(const f32x2*)(gbuf + m * GS + DETERD + j);
        f32x2 ug = *(const f32x2*)(gbuf + m * GS + 2 * DETERD + j);
        f32x2 dp = *(const f32x2*)(deter + m * DETERD + j);
        float u0 = sigm(ug.x + UPD_BIAS), u1 = sigm(ug.y + UPD_BIAS);
        float c0 = siluf(sigm(rg.x) * cg.x), c1 = siluf(sigm(rg.y) * cg.y);
        f32x2 dn;
        dn.x = u0 * c0 + (1.f - u0) * dp.x;
        dn.y = u1 * c1 + (1.f - u1) * dp.y;
        *(f32x2*)(deter + m * DETERD + j) = dn;
        *(unsigned*)(abuf + m * AS + j) = pack2bf(dn.x, dn.y);    // A for prior_out
        __builtin_nontemporal_store(dn, (f32x2*)(out + ob + 160 + j));  // deter_n
      }
    }
    __syncthreads();

    // ---- (4) prior_out + prior_stats ----
    wmma_gemm<DETERD>(Wt_po, HIDD / 16, abuf, gbuf, lane, wave);
    wprefetch(Wt_ps, 512 * 64 * 2, tid);
    row_stats(gbuf, HIDD, red, stats, tid);
    for (int m = 0; m < BM; m++) {
      const float mean = stats[2 * m], rstd = stats[2 * m + 1];
      for (int jj = tid; jj < HIDD / 2; jj += NTH) {
        const int j = jj * 2;
        f32x2 x = *(const f32x2*)(gbuf + m * GS + j);
        float y0 = siluf((x.x - mean) * rstd * g_po[j] + b_po[j]);
        float y1 = siluf((x.y - mean) * rstd * g_po[j + 1] + b_po[j + 1]);
        *(unsigned*)(abuf + m * AS + j) = pack2bf(y0, y1);        // hp
      }
    }
    __syncthreads();
    wmma_gemm<HIDD>(Wt_ps, (2 * STOCH) / 16, abuf, gbuf, lane, wave);
    wprefetch(Wt_pn, 576 * 512 * 2, tid);
    __syncthreads();
    for (int m = 0; m < BM; m++) {
      const size_t ob = ((size_t)(r0 + m) * TLEN + t) * OUTD;
      for (int j = tid; j < 2 * STOCH; j += NTH) {
        float v = gbuf[m * GS + j] + b_ps[j];
        if (j < STOCH)
          __builtin_nontemporal_store(v, out + ob + 96 + j);                      // p_mean
        else
          __builtin_nontemporal_store(softplusf(v) + MIN_STD, out + ob + 96 + j); // p_std
      }
    }
    __syncthreads();

    // ---- (5) post_net: hq = silu(LN([deter_n | obs] @ W_pn)) ----
    for (int m = 0; m < BM; m++) {
      for (int jj = tid; jj < DETERD / 2; jj += NTH) {
        const int j = jj * 2;
        f32x2 d = *(const f32x2*)(deter + m * DETERD + j);
        *(unsigned*)(abuf + m * AS + j) = pack2bf(d.x, d.y);
      }
      for (int j = tid; j < OBSD; j += NTH) {
        float o = __builtin_nontemporal_load(
            obs + ((size_t)(r0 + m) * TLEN + t) * OBSD + j);
        abuf[m * AS + DETERD + j] = f2bf(o);
      }
    }
    __syncthreads();
    wmma_gemm<DETERD + OBSD>(Wt_pn, HIDD / 16, abuf, gbuf, lane, wave);
    wprefetch(Wt_pt, 512 * 64 * 2, tid);
    row_stats(gbuf, HIDD, red, stats, tid);
    for (int m = 0; m < BM; m++) {
      const float mean = stats[2 * m], rstd = stats[2 * m + 1];
      for (int jj = tid; jj < HIDD / 2; jj += NTH) {
        const int j = jj * 2;
        f32x2 x = *(const f32x2*)(gbuf + m * GS + j);
        float y0 = siluf((x.x - mean) * rstd * g_pn[j] + b_pn[j]);
        float y1 = siluf((x.y - mean) * rstd * g_pn[j + 1] + b_pn[j + 1]);
        *(unsigned*)(abuf + m * AS + j) = pack2bf(y0, y1);        // hq
      }
    }
    __syncthreads();

    // ---- (6) post_stats; new stoch carry ----
    wmma_gemm<HIDD>(Wt_pt, (2 * STOCH) / 16, abuf, gbuf, lane, wave);
    __syncthreads();
    for (int m = 0; m < BM; m++) {
      const size_t ob = ((size_t)(r0 + m) * TLEN + t) * OUTD;
      for (int j = tid; j < 2 * STOCH; j += NTH) {
        float v = gbuf[m * GS + j] + b_pt[j];
        if (j < STOCH) {
          __builtin_nontemporal_store(v, out + ob + j);           // q_mean
          __builtin_nontemporal_store(v, out + ob + 64 + j);      // stoch_n (mode)
          stoch[m * STOCH + j] = v;
        } else {
          __builtin_nontemporal_store(softplusf(v) + MIN_STD, out + ob + 32 + (j - STOCH));
        }
      }
    }
    __syncthreads();
  }
}

// ---------------------------------------------------------------------------
// Host launch
// ---------------------------------------------------------------------------
extern "C" void kernel_launch(void* const* d_in, const int* in_sizes, int n_in,
                              void* d_out, int out_size, void* d_ws, size_t ws_size,
                              hipStream_t stream) {
  (void)in_sizes; (void)n_in; (void)out_size; (void)ws_size;

  const float* obs        = (const float*)d_in[0];
  const float* action     = (const float*)d_in[1];
  const int*   is_first   = (const int*)  d_in[2];
  const float* W_pi       = (const float*)d_in[3];
  const float* g_pi       = (const float*)d_in[4];
  const float* b_pi       = (const float*)d_in[5];
  const float* W_gru      = (const float*)d_in[6];
  const float* g_gru      = (const float*)d_in[7];
  const float* b_gru      = (const float*)d_in[8];
  const float* W_po       = (const float*)d_in[9];
  const float* g_po       = (const float*)d_in[10];
  const float* b_po       = (const float*)d_in[11];
  const float* W_ps       = (const float*)d_in[12];
  const float* b_ps       = (const float*)d_in[13];
  const float* W_pn       = (const float*)d_in[14];
  const float* g_pn       = (const float*)d_in[15];
  const float* b_pn       = (const float*)d_in[16];
  const float* W_pt       = (const float*)d_in[17];
  const float* b_pt       = (const float*)d_in[18];
  const float* init_deter = (const float*)d_in[19];
  float* out = (float*)d_out;

  // workspace carve (bytes, 256-aligned)
  char* ws = (char*)d_ws;
  unsigned short* Wt_pi  = (unsigned short*)(ws + 0);        // 512 x 64
  unsigned short* Wt_gru = (unsigned short*)(ws + 65536);    // 1536 x 1024
  unsigned short* Wt_po  = (unsigned short*)(ws + 3211264);  // 512 x 512
  unsigned short* Wt_ps  = (unsigned short*)(ws + 3735552);  // 64 x 512
  unsigned short* Wt_pn  = (unsigned short*)(ws + 3801088);  // 512 x 576
  unsigned short* Wt_pt  = (unsigned short*)(ws + 4390912);  // 64 x 512
  float* d_init          = (float*)(ws + 4456448);           // 512
  float* s_init          = (float*)(ws + 4458496);           // 32

  // weight transpose+bf16 conversion (weights stay L2-resident afterwards)
  k_wprep<<<128, 256, 0, stream>>>(W_pi, Wt_pi, 40, 64, 512);
  k_wprep<<<1024, 256, 0, stream>>>(W_gru, Wt_gru, 1024, 1024, 1536);
  k_wprep<<<512, 256, 0, stream>>>(W_po, Wt_po, 512, 512, 512);
  k_wprep<<<128, 256, 0, stream>>>(W_ps, Wt_ps, 512, 512, 64);
  k_wprep<<<512, 256, 0, stream>>>(W_pn, Wt_pn, 576, 576, 512);
  k_wprep<<<128, 256, 0, stream>>>(W_pt, Wt_pt, 512, 512, 64);

  k_init_state<<<1, 512, 0, stream>>>(init_deter, W_po, g_po, b_po, W_ps, b_ps,
                                      d_init, s_init);

  const size_t smem_bytes =
      (size_t)(BM * DETERD + BM * STOCH + BM * GS + 512 + 32) * sizeof(float) +
      (size_t)(BM * AS) * sizeof(unsigned short);   // ~168 KB of 320 KB WGP LDS
  hipFuncSetAttribute((const void*)k_rssm,
                      hipFuncAttributeMaxDynamicSharedMemorySize, (int)smem_bytes);

  k_rssm<<<BSZ / BM, NTH, smem_bytes, stream>>>(
      obs, action, is_first,
      g_pi, b_pi, g_gru, b_gru, g_po, b_po, b_ps, g_pn, b_pn, b_pt,
      Wt_pi, Wt_gru, Wt_po, Wt_ps, Wt_pn, Wt_pt,
      d_init, s_init, out);
}
